// QuantLlamaMLP_84954453115513
// MI455X (gfx1250) — matrix operand order repaired and checked by reading, not compiled, not run on hardware
//
#include <hip/hip_runtime.h>
#include <cstdint>

typedef __attribute__((ext_vector_type(16))) _Float16 v16h;
typedef __attribute__((ext_vector_type(8)))  _Float16 v8h;
typedef __attribute__((ext_vector_type(4)))  _Float16 v4h;
typedef __attribute__((ext_vector_type(2)))  _Float16 v2h;
typedef __attribute__((ext_vector_type(8)))  float    v8f;
typedef __attribute__((ext_vector_type(4)))  int      v4i;

#define BM 128      // block tile M
#define BN 64       // block tile N
#define BK 64       // block tile K (two wmma k=32 steps; always inside ONE AWQ group)
#define SA 72       // padded LDS stride (halves) for A rows   (144B = 9*16B)
#define SB 72       // padded LDS stride (halves) for B^T rows
#define NT 256      // 8 waves of 32

#if defined(__has_builtin)
#if __has_builtin(__builtin_amdgcn_global_load_async_to_lds_b128)
#define HAVE_ASYNC_LDS 1
#endif
#endif

#if defined(HAVE_ASYNC_LDS)
typedef __attribute__((address_space(1))) v4i* as1_v4i;
typedef __attribute__((address_space(3))) v4i* as3_v4i;
#endif

static __device__ __forceinline__ v16h cat8(v8h lo, v8h hi) {
  return __builtin_shufflevector(lo, hi, 0,1,2,3,4,5,6,7,8,9,10,11,12,13,14,15);
}
static __device__ __forceinline__ v4h cat2(v2h lo, v2h hi) {
  return __builtin_shufflevector(lo, hi, 0,1,2,3);
}

// Per-(group, packed-col) dequant constants: zh = packed halves (1024+z),
// s2 = packed f16 scales.  (q - z)*s == ((1024+q) - (1024+z)) * s, exact in f16.
struct DeqC { v2h zh[4]; v2h s2[4]; };

template<int N>
static __device__ __forceinline__ void build_consts(
    const int* __restrict__ qz, const float* __restrict__ sc,
    int grp, int cw, int ncol, DeqC& c) {
  const uint32_t z = (uint32_t)qz[(uint32_t)grp * (N >> 3) + (uint32_t)cw];
  const float* sp = sc + (uint32_t)grp * N + (uint32_t)ncol;
  const float4 sa = *(const float4*)sp;
  const float4 sb = *(const float4*)(sp + 4);
  const float sf[8] = {sa.x, sa.y, sa.z, sa.w, sb.x, sb.y, sb.z, sb.w};
  #pragma unroll
  for (int p = 0; p < 4; ++p) {
    const uint32_t zw = 0x64006400u | ((z >> (8 * p)) & 0xFu)
                                    | (((z >> (8 * p)) & 0xF0u) << 12);
    c.zh[p] = __builtin_bit_cast(v2h, zw);
    v2h s2; s2.x = (_Float16)sf[2 * p]; s2.y = (_Float16)sf[2 * p + 1];
    c.s2[p] = s2;
  }
}

// Dequant two adjacent k-rows (row0, row0+1) of one packed int32 column into
// transposed LDS tile BsT[n][k] — packed f16 math, half2 (k,k+1) stores.
template<int N>
static __device__ __forceinline__ void dequant_rows2(
    const int* __restrict__ qw, int row0, int cw,
    const DeqC& c, _Float16* __restrict__ BsT, int kk0, int cc) {
  const uint32_t qa = (uint32_t)qw[(uint32_t)row0 * (N >> 3) + (uint32_t)cw];
  const uint32_t qb = (uint32_t)qw[(uint32_t)(row0 + 1) * (N >> 3) + (uint32_t)cw];
  #pragma unroll
  for (int p = 0; p < 4; ++p) {
    const uint32_t wa = 0x64006400u | ((qa >> (8 * p)) & 0xFu)
                                    | (((qa >> (8 * p)) & 0xF0u) << 12);
    const uint32_t wb = 0x64006400u | ((qb >> (8 * p)) & 0xFu)
                                    | (((qb >> (8 * p)) & 0xF0u) << 12);
    const v2h va = (__builtin_bit_cast(v2h, wa) - c.zh[p]) * c.s2[p]; // cols 2p,2p+1 @ row0
    const v2h vb = (__builtin_bit_cast(v2h, wb) - c.zh[p]) * c.s2[p]; // cols 2p,2p+1 @ row0+1
    const v2h c0 = __builtin_shufflevector(va, vb, 0, 2);  // col 2p   : rows k,k+1
    const v2h c1 = __builtin_shufflevector(va, vb, 1, 3);  // col 2p+1 : rows k,k+1
    *(v2h*)&BsT[(cc * 8 + 2 * p    ) * SB + kk0] = c0;
    *(v2h*)&BsT[(cc * 8 + 2 * p + 1) * SB + kk0] = c1;
  }
}

// FUSED=true : h = f16( silu(x@Wg) * (x@Wu) )   (A = x, f32)
// FUSED=false: out_f32 = h @ Wd                  (A = h, f16, async-copied to LDS)
template<bool FUSED, int K, int N>
__global__ __launch_bounds__(NT)
void awq_mlp_gemm(const float*    __restrict__ Xf,
                  const _Float16* __restrict__ Xh,
                  const int*   __restrict__ qwG, const float* __restrict__ scG,
                  const int*   __restrict__ qzG,
                  const int*   __restrict__ qwU, const float* __restrict__ scU,
                  const int*   __restrict__ qzU,
                  _Float16* __restrict__ Hout, float* __restrict__ Fout) {
  __shared__ _Float16 As [BM * SA];
  __shared__ _Float16 BsG[BN * SB];                       // transposed: [n][k]
  __shared__ _Float16 BsU[(FUSED ? BN : 1) * SB];

  const int tid  = threadIdx.x;
  const int lane = tid & 31;
  const int wid  = tid >> 5;
  const int lr   = lane & 15;       // col / row selector within fragment
  const int lh   = lane >> 4;       // lane-half (K split per ISA layout)
  const int wm   = (wid & 3) << 5;  // wave M offset in block tile (0..96)
  const int wn   = (wid >> 2) << 5; // wave N offset (0 or 32)
  const int m0   = blockIdx.y * BM;
  const int n0   = blockIdx.x * BN;

  // dequant work mapping: thread -> (packed col cc, k-row pair kp)
  const int cc   = tid & 7;
  const int kk0  = (tid >> 3) << 1;       // 0,2,..,62
  const int cw   = (n0 >> 3) + cc;        // packed-int32 column (global)
  const int ncol = n0 + (cc << 3);        // first fp column of this word

  const float*    Xfb = FUSED ? Xf + (size_t)m0 * K : nullptr;
  const _Float16* Xhb = FUSED ? nullptr : Xh + (size_t)m0 * K;

  DeqC cG, cU;
  v8f accG[2][2] = {};
  v8f accU[2][2] = {};

  for (int k0 = 0; k0 < K; k0 += BK) {
    // AWQ group constants change only every 128 rows of K (2 K-tiles)
    if ((k0 & 127) == 0) {
      const int grp = k0 >> 7;
      build_consts<N>(qzG, scG, grp, cw, ncol, cG);
      if constexpr (FUSED) build_consts<N>(qzU, scU, grp, cw, ncol, cU);
    }

    // ---------------- stage A tile into LDS ----------------
    if constexpr (FUSED) {
      #pragma unroll
      for (int i = 0; i < 8; ++i) {
        const int idx = tid + i * NT;
        const int row = idx >> 4;
        const int kc  = (idx & 15) << 2;
        const float4 v = *(const float4*)(Xfb + (uint32_t)(row * K + k0 + kc));
        const v2h h01 = __builtin_bit_cast(v2h, __builtin_amdgcn_cvt_pkrtz(v.x, v.y));
        const v2h h23 = __builtin_bit_cast(v2h, __builtin_amdgcn_cvt_pkrtz(v.z, v.w));
        *(v4h*)&As[row * SA + kc] = cat2(h01, h23);
      }
    } else {
      #pragma unroll
      for (int i = 0; i < 4; ++i) {
        const int idx = tid + i * NT;
        const int row = idx >> 3;
        const int kc  = (idx & 7) << 3;
#if defined(HAVE_ASYNC_LDS)
        // CDNA5 async DMA: global -> LDS without VGPR staging (ASYNCcnt)
        __builtin_amdgcn_global_load_async_to_lds_b128(
            (as1_v4i)(Xhb + (uint32_t)(row * K + k0 + kc)),
            (as3_v4i)&As[row * SA + kc],
            0, 0);
#else
        *(v8h*)&As[row * SA + kc] =
            *(const v8h*)(Xhb + (uint32_t)(row * K + k0 + kc));
#endif
      }
    }

    // ------------- dequant int4 B tile(s) -> LDS (transposed) -------------
    dequant_rows2<N>(qwG, k0 + kk0, cw, cG, BsG, kk0, cc);
    if (k0 + BK < K)
      __builtin_prefetch(qwG + (uint32_t)(k0 + BK + kk0) * (N >> 3) + cw, 0, 1);
    if constexpr (FUSED) {
      dequant_rows2<N>(qwU, k0 + kk0, cw, cU, BsU, kk0, cc);
      if (k0 + BK < K)
        __builtin_prefetch(qwU + (uint32_t)(k0 + BK + kk0) * (N >> 3) + cw, 0, 1);
    }

#if defined(HAVE_ASYNC_LDS)
    if constexpr (!FUSED) {
#if __has_builtin(__builtin_amdgcn_s_wait_asynccnt)
      __builtin_amdgcn_s_wait_asynccnt(0);
#else
      asm volatile("s_wait_asynccnt 0" ::: "memory");
#endif
    }
#endif
    __syncthreads();

    // ---------------- WMMA compute on the tile ----------------
    #pragma unroll
    for (int kk = 0; kk < BK; kk += 32) {
      v16h a[2];
      #pragma unroll
      for (int tm = 0; tm < 2; ++tm) {
        // A 16x32 f16: lanes 0-15 hold K 0..15, lanes 16-31 K 16..31
        const _Float16* ap = &As[(wm + tm * 16 + lr) * SA + kk + lh * 16];
        a[tm] = cat8(*(const v8h*)ap, *(const v8h*)(ap + 8));
      }
      #pragma unroll
      for (int tn = 0; tn < 2; ++tn) {
        // B 32x16 f16: lanes 0-15 hold K 0..15, lanes 16-31 K 16..31 (col = lane%16)
        const _Float16* bp = &BsG[(wn + tn * 16 + lr) * SB + kk + (lh << 4)];
        const v16h b = cat8(*(const v8h*)bp, *(const v8h*)(bp + 8));
        #pragma unroll
        for (int tm = 0; tm < 2; ++tm)
          accG[tm][tn] = __builtin_amdgcn_wmma_f32_16x16x32_f16(
              false, a[tm], false, b, (short)0, accG[tm][tn], false, false);
      }
      if constexpr (FUSED) {
        #pragma unroll
        for (int tn = 0; tn < 2; ++tn) {
          const _Float16* bp = &BsU[(wn + tn * 16 + lr) * SB + kk + (lh << 4)];
          const v16h b = cat8(*(const v8h*)bp, *(const v8h*)(bp + 8));
          #pragma unroll
          for (int tm = 0; tm < 2; ++tm)
            accU[tm][tn] = __builtin_amdgcn_wmma_f32_16x16x32_f16(
                false, a[tm], false, b, (short)0, accU[tm][tn], false, false);
        }
      }
    }
    __syncthreads();
  }

  // ---------------- epilogue ----------------
  // C/D 16x16 f32 layout: VGPR r -> row (r + 8*lane_half), col = lane%16
  const int mb = m0 + wm + lh * 8;
  const int nb = n0 + wn + lr;
  #pragma unroll
  for (int tm = 0; tm < 2; ++tm)
    #pragma unroll
    for (int tn = 0; tn < 2; ++tn)
      #pragma unroll
      for (int r = 0; r < 8; ++r) {
        const uint32_t o = (uint32_t)(mb + tm * 16 + r) * (uint32_t)N
                         + (uint32_t)(nb + tn * 16);
        if constexpr (FUSED) {
          const float g = accG[tm][tn][r];
          const float u = accU[tm][tn][r];
          const float s = g / (1.0f + __expf(-g));   // SiLU
          Hout[o] = (_Float16)(s * u);
        } else {
          Fout[o] = accG[tm][tn][r];
        }
      }
}

extern "C" void kernel_launch(void* const* d_in, const int* in_sizes, int n_in,
                              void* d_out, int out_size, void* d_ws, size_t ws_size,
                              hipStream_t stream) {
  const float* x    = (const float*)d_in[0];
  const int*   g_qw = (const int*)  d_in[1];
  const float* g_sc = (const float*)d_in[2];
  const int*   g_qz = (const int*)  d_in[3];
  const int*   u_qw = (const int*)  d_in[4];
  const float* u_sc = (const float*)d_in[5];
  const int*   u_qz = (const int*)  d_in[6];
  const int*   d_qw = (const int*)  d_in[7];
  const float* d_sc = (const float*)d_in[8];
  const int*   d_qz = (const int*)  d_in[9];
  float*       out  = (float*)d_out;
  _Float16*    h    = (_Float16*)d_ws;   // [8192, 11008] f16 intermediate

  constexpr int Mtok = 8192, Din = 4096, Dmid = 11008, Dout2 = 4096;
  (void)in_sizes; (void)n_in; (void)out_size; (void)ws_size;

  dim3 block(NT);
  // GEMM1: h = f16( silu(x@Wg) * (x@Wu) )
  dim3 grid1(Dmid / BN, Mtok / BM);
  awq_mlp_gemm<true, Din, Dmid><<<grid1, block, 0, stream>>>(
      x, nullptr, g_qw, g_sc, g_qz, u_qw, u_sc, u_qz, h, nullptr);
  // GEMM2: out = h @ Wd
  dim3 grid2(Dout2 / BN, Mtok / BM);
  awq_mlp_gemm<false, Dmid, Dout2><<<grid2, block, 0, stream>>>(
      nullptr, h, d_qw, d_sc, d_qz, nullptr, nullptr, nullptr, nullptr, out);
}
